// FixedSplitAttention_60163901882945
// MI455X (gfx1250) — matrix operand order repaired and checked by reading, not compiled, not run on hardware
//
#include <hip/hip_runtime.h>

// ---------------------------------------------------------------------------
// FixedSplitAttention for MI455X (gfx1250, wave32, WMMA bf16 16x16x32)
// B=2 S=2048 HID=2048 H=16 HKV=4 D=128 SPLIT=64 C=32
// Pipeline: QKV proj GEMMs (fp32 in -> bf16 ws), bf16 chunked attention with
// async-to-LDS tile fills (GLOBAL_LOAD_ASYNC_TO_LDS_B128, ASYNCcnt),
// output proj GEMM (bf16 ws -> fp32 out).
// ---------------------------------------------------------------------------

typedef __attribute__((ext_vector_type(16))) __bf16 v16bf;
typedef __attribute__((ext_vector_type(8)))  float  v8f;

union Frag { v16bf v; unsigned u[8]; };

__device__ __forceinline__ unsigned short f2bf(float f) {
    union { float f; unsigned u; } x; x.f = f;
    unsigned r = x.u + 0x7FFFu + ((x.u >> 16) & 1u);   // round-to-nearest-even
    return (unsigned short)(r >> 16);
}

__device__ __forceinline__ unsigned pack2(float a, float b) {
#if __has_builtin(__builtin_amdgcn_cvt_pk_bf16_f32)
    auto r = __builtin_amdgcn_cvt_pk_bf16_f32(a, b);   // v_cvt_pk_bf16_f32
    unsigned u; __builtin_memcpy(&u, &r, 4); return u;
#else
    return (unsigned)f2bf(a) | ((unsigned)f2bf(b) << 16);
#endif
}

__device__ __forceinline__ v8f wmma_bf16(const v16bf& a, const v16bf& b, v8f c) {
    return __builtin_amdgcn_wmma_f32_16x16x32_bf16(
        false, a, false, b, (short)0, c, false, false);
}

__device__ __forceinline__ v8f vzero8() {
    v8f z;
    #pragma unroll
    for (int i = 0; i < 8; ++i) z[i] = 0.0f;
    return z;
}

// ---- gfx1250 async global->LDS copy (16B per lane), ASYNCcnt tracked ------
// GV mode: vdst = LDS byte address (low 32 bits of flat pointer), vaddr = u64.
__device__ __forceinline__ void async_cp16(const void* g, void* l) {
    unsigned lds = (unsigned)(unsigned long long)l;
    asm volatile("global_load_async_to_lds_b128 %0, %1, off"
                 :: "v"(lds), "v"((unsigned long long)g)
                 : "memory");
}

__device__ __forceinline__ void wait_async_lds() {
    asm volatile("s_wait_asynccnt 0x0" ::: "memory");
}

// ---------------------------------------------------------------------------
// GEMM: C = A[M,K] @ W[N,K]^T, bf16 WMMA math, f32 accumulate.
//   ABF16: A operand is bf16 (else fp32)
//   OMODE: 0 = fp32 row-major, 1 = bf16 row-major,
//          2 = bf16 "V-transposed": out[((b*4+hkv)*128+d)*2048 + s]
// block: 256 threads / 8 waves, tile 128x128, K-step 32,
// wave tile 64x32 -> 4x2 accum fragments.
// ---------------------------------------------------------------------------
#define LDSTRIDE 36   // ushorts/row: 32 used + pad, conflict-free, 8B rows

template<bool ABF16, int OMODE>
__global__ __launch_bounds__(256) void gemm_kernel(
    const void* __restrict__ Ap, const float* __restrict__ W,
    void* __restrict__ Cout, int M, int N, int K)
{
    __shared__ unsigned short As[128 * LDSTRIDE];
    __shared__ unsigned short Ws[128 * LDSTRIDE];

    const int tid  = threadIdx.x;
    const int lane = tid & 31;
    const int wid  = tid >> 5;
    const int mBase = blockIdx.y * 128;
    const int nBase = blockIdx.x * 128;
    const int wm0 = (wid & 1) * 64;
    const int wn0 = (wid >> 1) * 32;
    const int l15 = lane & 15;
    const int hiA = (lane >> 4) * 8;    // A-frag K shift (lanes 16-31: +8)
    const int khB = (lane >> 4) * 8;    // B-frag K shift (lanes 16-31: +8)
    const int hi8 = (lane >> 4) * 8;    // C-frag row shift

    v8f acc[4][2];
    #pragma unroll
    for (int mf = 0; mf < 4; ++mf)
        #pragma unroll
        for (int nf = 0; nf < 2; ++nf) acc[mf][nf] = vzero8();

    for (int k0 = 0; k0 < K; k0 += 32) {
        __syncthreads();
        if (ABF16) {
            const unsigned short* A = (const unsigned short*)Ap;
            #pragma unroll
            for (int i = 0; i < 4; ++i) {
                int u = tid + 256 * i;        // 1024 uint2 units (4 bf16)
                int row = u >> 3, c4 = (u & 7) << 2;
                uint2 d = *(const uint2*)(A + (size_t)(mBase + row) * K + k0 + c4);
                *(uint2*)&As[row * LDSTRIDE + c4] = d;
            }
        } else {
            const float* A = (const float*)Ap;
            #pragma unroll
            for (int i = 0; i < 4; ++i) {
                int u = tid + 256 * i;        // 1024 float4 units
                int row = u >> 3, c4 = (u & 7) << 2;
                float4 fa = *(const float4*)(A + (size_t)(mBase + row) * K + k0 + c4);
                uint2 pa; pa.x = pack2(fa.x, fa.y); pa.y = pack2(fa.z, fa.w);
                *(uint2*)&As[row * LDSTRIDE + c4] = pa;
            }
        }
        #pragma unroll
        for (int i = 0; i < 4; ++i) {
            int u = tid + 256 * i;
            int row = u >> 3, c4 = (u & 7) << 2;
            float4 fw = *(const float4*)(W + (size_t)(nBase + row) * K + k0 + c4);
            uint2 pw; pw.x = pack2(fw.x, fw.y); pw.y = pack2(fw.z, fw.w);
            *(uint2*)&Ws[row * LDSTRIDE + c4] = pw;
        }
        __syncthreads();

        Frag bfr[2];
        #pragma unroll
        for (int nf = 0; nf < 2; ++nf) {
            const unsigned short* p = &Ws[(wn0 + nf * 16 + l15) * LDSTRIDE + khB];
            #pragma unroll
            for (int i = 0; i < 8; ++i) bfr[nf].u[i] = *(const unsigned*)(p + 2 * (i & 3) + ((i >> 2) << 4));
        }
        #pragma unroll
        for (int mf = 0; mf < 4; ++mf) {
            Frag af;
            const unsigned short* p = &As[(wm0 + mf * 16 + l15) * LDSTRIDE + hiA];
            #pragma unroll
            for (int i = 0; i < 4; ++i) af.u[i]     = *(const unsigned*)(p + 2 * i);
            #pragma unroll
            for (int i = 0; i < 4; ++i) af.u[4 + i] = *(const unsigned*)(p + 16 + 2 * i);
            acc[mf][0] = wmma_bf16(af.v, bfr[0].v, acc[mf][0]);
            acc[mf][1] = wmma_bf16(af.v, bfr[1].v, acc[mf][1]);
        }
    }

    if (OMODE == 0) {
        float* C = (float*)Cout;
        #pragma unroll
        for (int mf = 0; mf < 4; ++mf)
            #pragma unroll
            for (int nf = 0; nf < 2; ++nf)
                #pragma unroll
                for (int j = 0; j < 8; ++j) {
                    int row = mBase + wm0 + mf * 16 + hi8 + j;
                    int col = nBase + wn0 + nf * 16 + l15;
                    C[(size_t)row * N + col] = acc[mf][nf][j];
                }
    } else if (OMODE == 1) {
        unsigned short* C = (unsigned short*)Cout;
        #pragma unroll
        for (int mf = 0; mf < 4; ++mf)
            #pragma unroll
            for (int nf = 0; nf < 2; ++nf)
                #pragma unroll
                for (int j = 0; j < 8; ++j) {
                    int row = mBase + wm0 + mf * 16 + hi8 + j;
                    int col = nBase + wn0 + nf * 16 + l15;
                    C[(size_t)row * N + col] = f2bf(acc[mf][nf][j]);
                }
    } else {  // OMODE==2: V^T per (b, kv-head): [((b*4+hkv)*128+d)*2048 + s]
        unsigned short* C = (unsigned short*)Cout;
        #pragma unroll
        for (int mf = 0; mf < 4; ++mf)
            #pragma unroll
            for (int nf = 0; nf < 2; ++nf) {
                int n   = nBase + wn0 + nf * 16 + l15;   // n = hkv*128 + d
                int hkv = n >> 7, d = n & 127;
                int s0  = mBase + wm0 + mf * 16 + hi8;   // 8 consecutive keys
                int b   = s0 >> 11, s = s0 & 2047;
                uint4 pk;
                pk.x = pack2(acc[mf][nf][0], acc[mf][nf][1]);
                pk.y = pack2(acc[mf][nf][2], acc[mf][nf][3]);
                pk.z = pack2(acc[mf][nf][4], acc[mf][nf][5]);
                pk.w = pack2(acc[mf][nf][6], acc[mf][nf][7]);
                *(uint4*)&C[((size_t)((b * 4 + hkv) * 128 + d)) * 2048 + s] = pk;
            }
    }
}

// ---------------------------------------------------------------------------
// Attention: fixed 64-wide KV chunks, per-chunk softmax, mean over 32 chunks.
// Inputs bf16: Q row-major (4096,2048), K row-major (4096,512),
//              Vt transposed ((b*4+hkv)*128+d, 2048).
// block = 128 threads / 4 waves; wave = 16 q rows; grid = (S/64, H, B).
// ---------------------------------------------------------------------------
#define QSTR 136   // Q/K LDS row stride (ushorts): 272B = 17*16, b128-aligned
#define VSTR 88    // Vt LDS row stride (ushorts): 176B = 11*16, b128-aligned
#define PSTR 72    // P LDS row stride (ushorts)
#define ATT_SCALE 0.08838834764831845f

__global__ __launch_bounds__(128) void attn_split(
    const unsigned short* __restrict__ Q, const unsigned short* __restrict__ Kp,
    const unsigned short* __restrict__ Vtg, const float* __restrict__ Mask,
    unsigned short* __restrict__ Out)
{
    __shared__ unsigned short Qs[64 * QSTR];     // 64 q rows x 128 d
    __shared__ unsigned short Ks[64 * QSTR];     // 64 keys  x 128 d
    __shared__ unsigned short Vt[128 * VSTR];    // 128 d x 64 keys
    __shared__ unsigned short Ps[4][16 * PSTR];  // per-wave 16 x 64 probs

    const int tid = threadIdx.x, lane = tid & 31, wid = tid >> 5;
    const int qt = blockIdx.x, h = blockIdx.y, b = blockIdx.z;
    const int hkv = h >> 2;
    const int qBase = qt * 64;
    const int l15 = lane & 15;
    const int hiA = (lane >> 4) * 8;
    const int khB = (lane >> 4) * 8;
    const int hi8 = (lane >> 4) * 8;

    // ---- Q tile (64 x 128 bf16) -> LDS, async b128 ----
    {
        const unsigned short* src = Q + ((size_t)(b * 2048 + qBase)) * 2048 + h * 128;
        #pragma unroll
        for (int i = 0; i < 8; ++i) {
            int u = tid + 128 * i;               // 1024 b128 units
            int row = u >> 4, c8 = (u & 15) << 3;
            async_cp16(src + (size_t)row * 2048 + c8, &Qs[row * QSTR + c8]);
        }
        wait_async_lds();
    }
    __syncthreads();

    // ---- prebuild Q A-fragments: 4 K-steps of 32 over D=128 ----
    Frag qa[4];
    #pragma unroll
    for (int kb = 0; kb < 4; ++kb) {
        const unsigned short* p = &Qs[(wid * 16 + l15) * QSTR + kb * 32 + hiA];
        #pragma unroll
        for (int i = 0; i < 4; ++i) qa[kb].u[i]     = *(const unsigned*)(p + 2 * i);
        #pragma unroll
        for (int i = 0; i < 4; ++i) qa[kb].u[4 + i] = *(const unsigned*)(p + 16 + 2 * i);
    }

    v8f o[8];
    #pragma unroll
    for (int i = 0; i < 8; ++i) o[i] = vzero8();

    const float* maskBase = Mask + (size_t)b * 2048 * 2048;

    for (int c = 0; c < 32; ++c) {
        __syncthreads();   // all waves done reading previous chunk's K/V
        // ---- async fill K (64x128) and Vt (128x64) bf16 tiles ----
        const unsigned short* ksrc = Kp + ((size_t)(b * 2048 + c * 64)) * 512 + hkv * 128;
        const unsigned short* vsrc = Vtg + ((size_t)((b * 4 + hkv) * 128)) * 2048 + c * 64;
        #pragma unroll
        for (int i = 0; i < 8; ++i) {
            int u = tid + 128 * i;
            int row = u >> 4, c8 = (u & 15) << 3;
            async_cp16(ksrc + (size_t)row * 512 + c8, &Ks[row * QSTR + c8]);
        }
        #pragma unroll
        for (int i = 0; i < 8; ++i) {
            int u = tid + 128 * i;               // 128 d rows x 8 units
            int row = u >> 3, c8 = (u & 7) << 3;
            async_cp16(vsrc + (size_t)row * 2048 + c8, &Vt[row * VSTR + c8]);
        }
        wait_async_lds();
        __syncthreads();

        // ---- S(16x64) = Q(16x128) @ K^T(128x64) ----
        v8f s[4];
        #pragma unroll
        for (int nt = 0; nt < 4; ++nt) s[nt] = vzero8();
        #pragma unroll
        for (int kb = 0; kb < 4; ++kb) {
            #pragma unroll
            for (int nt = 0; nt < 4; ++nt) {
                Frag bf;
                const unsigned short* p = &Ks[(nt * 16 + l15) * QSTR + kb * 32 + khB];
                #pragma unroll
                for (int i = 0; i < 8; ++i) bf.u[i] = *(const unsigned*)(p + 2 * (i & 3) + ((i >> 2) << 4));
                s[nt] = wmma_bf16(qa[kb].v, bf.v, s[nt]);
            }
        }

        // ---- per-row softmax over the 64-key chunk ----
        #pragma unroll
        for (int j = 0; j < 8; ++j) {
            int qrow = qBase + wid * 16 + hi8 + j;
            const float* mrow = maskBase + (size_t)qrow * 2048 + c * 64;
            float x[4];
            #pragma unroll
            for (int nt = 0; nt < 4; ++nt)
                x[nt] = s[nt][j] * ATT_SCALE + mrow[nt * 16 + l15];
            float mx = fmaxf(fmaxf(x[0], x[1]), fmaxf(x[2], x[3]));
            #pragma unroll
            for (int off = 8; off >= 1; off >>= 1)
                mx = fmaxf(mx, __shfl_xor(mx, off, 32));
            float r = 0.0f;
            #pragma unroll
            for (int nt = 0; nt < 4; ++nt) { x[nt] = __expf(x[nt] - mx); r += x[nt]; }
            #pragma unroll
            for (int off = 8; off >= 1; off >>= 1)
                r += __shfl_xor(r, off, 32);
            float inv = 1.0f / r;
            #pragma unroll
            for (int nt = 0; nt < 4; ++nt)
                Ps[wid][(hi8 + j) * PSTR + nt * 16 + l15] = f2bf(x[nt] * inv);
        }
        // per-wave LDS region: same-wave DS ops are in order -> no barrier

        // ---- O(16x128) += P(16x64) @ V(64x128) ----
        #pragma unroll
        for (int kb = 0; kb < 2; ++kb) {
            Frag pa;
            const unsigned short* pp = &Ps[wid][l15 * PSTR + kb * 32 + hiA];
            #pragma unroll
            for (int i = 0; i < 4; ++i) pa.u[i]     = *(const unsigned*)(pp + 2 * i);
            #pragma unroll
            for (int i = 0; i < 4; ++i) pa.u[4 + i] = *(const unsigned*)(pp + 16 + 2 * i);
            #pragma unroll
            for (int nt = 0; nt < 8; ++nt) {
                Frag vb;
                const unsigned short* vp = &Vt[(nt * 16 + l15) * VSTR + kb * 32 + khB];
                #pragma unroll
                for (int i = 0; i < 8; ++i) vb.u[i] = *(const unsigned*)(vp + 2 * (i & 3) + ((i >> 2) << 4));
                o[nt] = wmma_bf16(pa.v, vb.v, o[nt]);
            }
        }
    }

    // combined / (total_weight + eps); per-chunk wsum==1 => total = 32
    const float fin = 1.0f / (32.0f + 1e-9f);
    #pragma unroll
    for (int nt = 0; nt < 8; ++nt)
        #pragma unroll
        for (int j = 0; j < 8; ++j) {
            int qrow = qBase + wid * 16 + hi8 + j;
            int d = nt * 16 + l15;
            Out[((size_t)(b * 2048 + qrow)) * 2048 + h * 128 + d] =
                f2bf(o[nt][j] * fin);
        }
}

// ---------------------------------------------------------------------------
// launch
// ---------------------------------------------------------------------------
extern "C" void kernel_launch(void* const* d_in, const int* in_sizes, int n_in,
                              void* d_out, int out_size, void* d_ws, size_t ws_size,
                              hipStream_t stream)
{
    const float* X    = (const float*)d_in[0];   // (2,2048,2048)
    const float* Mask = (const float*)d_in[1];   // (2,1,2048,2048)
    const float* Wq   = (const float*)d_in[2];   // (2048,2048)
    const float* Wk   = (const float*)d_in[3];   // (512,2048)
    const float* Wv   = (const float*)d_in[4];   // (512,2048)
    const float* Wo   = (const float*)d_in[5];   // (2048,2048)
    float* out = (float*)d_out;                  // (2,2048,2048)

    unsigned short* Qw  = (unsigned short*)d_ws;          // 4096 x 2048 bf16
    unsigned short* Kw  = Qw  + (size_t)4096 * 2048;      // 4096 x 512  bf16
    unsigned short* Vtw = Kw  + (size_t)4096 * 512;       // (2*4*128) x 2048 bf16 (V^T)
    unsigned short* Aw  = Vtw + (size_t)4096 * 512;       // 4096 x 2048 bf16

    const dim3 blk(256);
    gemm_kernel<false, 1><<<dim3(16, 32), blk, 0, stream>>>(X, Wq, Qw, 4096, 2048, 2048);
    gemm_kernel<false, 1><<<dim3(4, 32),  blk, 0, stream>>>(X, Wk, Kw, 4096, 512, 2048);
    gemm_kernel<false, 2><<<dim3(4, 32),  blk, 0, stream>>>(X, Wv, Vtw, 4096, 512, 2048);

    attn_split<<<dim3(32, 16, 2), dim3(128), 0, stream>>>(Qw, Kw, Vtw, Mask, Aw);

    gemm_kernel<true, 0><<<dim3(16, 32), blk, 0, stream>>>(Aw, Wo, out, 4096, 2048, 2048);
}
